// MHTempCtxAttention_70042326663624
// MI455X (gfx1250) — compile-verified, matrix operand order checked by reading, not run on hardware
//
#include <hip/hip_runtime.h>

// ---------------------------------------------------------------------------
// MHTempCtxAttention for MI455X (gfx1250, wave32, WMMA + async LDS DMA)
// Pipeline:
//   1) one-pass fp32->bf16 conversion of activations & weights (HBM-bound, ~2us)
//   2) bf16 WMMA GEMMs (compute-bound part), tiles staged into LDS with
//      GLOBAL_LOAD_ASYNC_TO_LDS_B128 + s_wait_asynccnt double-buffer pipeline
//   3) rank-1 softmax attention on fp32 VALU (v_exp_f32), ctx emitted as bf16
//   4) output projection GEMM -> fp32 d_out
// ---------------------------------------------------------------------------

typedef __attribute__((ext_vector_type(16))) __bf16 v16bf;
typedef __attribute__((ext_vector_type(8)))  float  v8f;
typedef __attribute__((ext_vector_type(4)))  int    v4i;

#define BM   128      // block tile rows
#define BN   128      // block tile cols
#define BK   32       // K step (WMMA K for bf16)
#define LSTR 40       // LDS row stride in bf16 elems (32 + 8 pad; rows stay 16B aligned)

#if __has_builtin(__builtin_amdgcn_global_load_async_to_lds_b128)
#define ASYNC_LDS 1
#else
#define ASYNC_LDS 0
#endif

#if __has_builtin(__builtin_amdgcn_s_wait_asynccnt)
#define WAIT_ASYNC(n) __builtin_amdgcn_s_wait_asynccnt(n)
#else
#define WAIT_ASYNC(n) asm volatile("s_wait_asynccnt %0" :: "n"(n) : "memory")
#endif

typedef __attribute__((address_space(1))) v4i as1_v4i;   // global 16B vector
typedef __attribute__((address_space(3))) v4i as3_v4i;   // LDS 16B vector

__device__ __forceinline__ unsigned short f2bf(float f) {
    unsigned int u = __float_as_uint(f);
    u += 0x7FFFu + ((u >> 16) & 1u);           // round-to-nearest-even
    return (unsigned short)(u >> 16);
}

#if ASYNC_LDS
// 16B async DMA: LDS[dst] = MEM[src], tracked by ASYNCcnt (no VGPR round-trip)
__device__ __forceinline__ void async_cp16(const void* gsrc, void* ldsdst) {
    __builtin_amdgcn_global_load_async_to_lds_b128(
        (as1_v4i*)(unsigned long long)gsrc,
        (as3_v4i*)(unsigned int)(unsigned long long)ldsdst,
        /*offset=*/0, /*cpol=*/0);
}
#endif

// Load one 16x32 bf16 fragment from LDS (ISA striping):
//   lanes 0-15 : row = lane,    K = {0..7, 16..23}
//   lanes 16-31: row = lane-16, K = {8..15, 24..31}
__device__ __forceinline__ v16bf load_frag(const unsigned short* lds,
                                           int rowBase, int lane) {
    const int r  = lane & 15;
    const int kb = (lane >> 4) << 3;
    const unsigned short* p = lds + (rowBase + r) * LSTR + kb;
    union { v16bf v; uint4 q[2]; } u;
    u.q[0] = *reinterpret_cast<const uint4*>(p);
    u.q[1] = *reinterpret_cast<const uint4*>(p + 16);
    return u.v;
}

// C[M x N] = A[M x K](bf16) * W[N x K](bf16)^T + bias[N](f32), C = f32
__global__ void __launch_bounds__(256)
gemm_bias_bf16wmma(const unsigned short* __restrict__ A,
                   const unsigned short* __restrict__ W,
                   const float* __restrict__ bias,
                   float* __restrict__ C,
                   int Mrows, int Nout, int Kdim)
{
    __shared__ __align__(16) unsigned short As[2][BM * LSTR];
    __shared__ __align__(16) unsigned short Bs[2][BN * LSTR];

    const int tid  = threadIdx.x;
    const int lane = tid & 31;
    const int wave = tid >> 5;          // 0..7
    const int wr   = wave >> 2;         // 0..1 : wave row (64 rows each)
    const int wc   = wave & 3;          // 0..3 : wave col (32 cols each)

    const int blockRow = blockIdx.y * BM;
    const int blockCol = blockIdx.x * BN;

    v8f acc[4][2];
    #pragma unroll
    for (int i = 0; i < 4; ++i)
        #pragma unroll
        for (int j = 0; j < 2; ++j)
            acc[i][j] = (v8f){0.f, 0.f, 0.f, 0.f, 0.f, 0.f, 0.f, 0.f};

    const int nTiles = Kdim / BK;       // 32

#if ASYNC_LDS
    // ---- async double-buffered pipeline: 4 x b128 per thread per tile ----
    // chunk mapping: tile = A[128x32] ++ B[128x32] bf16 = 1024 x 16B chunks
    auto issue_tile = [&](int buf, int k0) {
        #pragma unroll
        for (int i = 0; i < 2; ++i) {                  // A half: chunks 0..511
            const int c   = tid + i * 256;
            const int row = c >> 2, seg = c & 3;       // 4 x 16B per row
            async_cp16(A + (size_t)(blockRow + row) * Kdim + k0 + seg * 8,
                       &As[buf][row * LSTR + seg * 8]);
        }
        #pragma unroll
        for (int i = 0; i < 2; ++i) {                  // B half
            const int c   = tid + i * 256;
            const int row = c >> 2, seg = c & 3;
            async_cp16(W + (size_t)(blockCol + row) * Kdim + k0 + seg * 8,
                       &Bs[buf][row * LSTR + seg * 8]);
        }
    };

    issue_tile(0, 0);
    for (int t = 0; t < nTiles; ++t) {
        const int cur = t & 1;
        if (t + 1 < nTiles) {
            issue_tile(cur ^ 1, (t + 1) * BK);         // overlap next tile DMA
            WAIT_ASYNC(4);                             // tile t complete (in-order)
        } else {
            WAIT_ASYNC(0);
        }
        __syncthreads();

        v16bf af[4], bfd[2];
        #pragma unroll
        for (int i = 0; i < 4; ++i) af[i]  = load_frag(As[cur], wr * 64 + i * 16, lane);
        #pragma unroll
        for (int j = 0; j < 2; ++j) bfd[j] = load_frag(Bs[cur], wc * 32 + j * 16, lane);

        #pragma unroll
        for (int i = 0; i < 4; ++i)
            #pragma unroll
            for (int j = 0; j < 2; ++j)
                acc[i][j] = __builtin_amdgcn_wmma_f32_16x16x32_bf16(
                    false, af[i], false, bfd[j], (short)0, acc[i][j], false, false);
        __syncthreads();                                // buf[cur] reusable
    }
#else
    // ---- sync fallback: batched wide loads, then wide LDS stores ----
    for (int t = 0; t < nTiles; ++t) {
        const int k0 = t * BK;
        uint4 ra[2], rb[2];
        #pragma unroll
        for (int i = 0; i < 2; ++i) {
            const int c = tid + i * 256, row = c >> 2, seg = c & 3;
            ra[i] = *reinterpret_cast<const uint4*>(
                A + (size_t)(blockRow + row) * Kdim + k0 + seg * 8);
            rb[i] = *reinterpret_cast<const uint4*>(
                W + (size_t)(blockCol + row) * Kdim + k0 + seg * 8);
        }
        __syncthreads();
        #pragma unroll
        for (int i = 0; i < 2; ++i) {
            const int c = tid + i * 256, row = c >> 2, seg = c & 3;
            *reinterpret_cast<uint4*>(&As[0][row * LSTR + seg * 8]) = ra[i];
            *reinterpret_cast<uint4*>(&Bs[0][row * LSTR + seg * 8]) = rb[i];
        }
        __syncthreads();

        v16bf af[4], bfd[2];
        #pragma unroll
        for (int i = 0; i < 4; ++i) af[i]  = load_frag(As[0], wr * 64 + i * 16, lane);
        #pragma unroll
        for (int j = 0; j < 2; ++j) bfd[j] = load_frag(Bs[0], wc * 32 + j * 16, lane);

        #pragma unroll
        for (int i = 0; i < 4; ++i)
            #pragma unroll
            for (int j = 0; j < 2; ++j)
                acc[i][j] = __builtin_amdgcn_wmma_f32_16x16x32_bf16(
                    false, af[i], false, bfd[j], (short)0, acc[i][j], false, false);
    }
#endif

    // ---- store: VGPR v -> row v (+8 for upper lane half), col = lane&15 ----
    const int cr = (lane >> 4) * 8;
    const int cc = lane & 15;
    #pragma unroll
    for (int i = 0; i < 4; ++i) {
        #pragma unroll
        for (int j = 0; j < 2; ++j) {
            const int col = blockCol + wc * 32 + j * 16 + cc;
            const float b = bias[col];
            #pragma unroll
            for (int v = 0; v < 8; ++v) {
                const int row = blockRow + wr * 64 + i * 16 + cr + v;
                C[(size_t)row * Nout + col] = acc[i][j][v] + b;
            }
        }
    }
}

// fp32 -> bf16 one-pass conversion (vectorized), n4 = elems/4
__global__ void __launch_bounds__(256)
cvt_f32_bf16(const float* __restrict__ src, unsigned short* __restrict__ dst, int n4)
{
    const int i = blockIdx.x * 256 + threadIdx.x;
    if (i < n4) {
        const float4 f = reinterpret_cast<const float4*>(src)[i];
        ushort4 h;
        h.x = f2bf(f.x); h.y = f2bf(f.y); h.z = f2bf(f.z); h.w = f2bf(f.w);
        reinterpret_cast<ushort4*>(dst)[i] = h;
    }
}

// Rank-1 per-token attention: scores[d][q] = q_d*k_q/8, so softmax row d is
// softmax_q(q_d*k_q): never materialize the 64x64 matrix. ctx emitted bf16.
__global__ void __launch_bounds__(256)
attn_rank1_softmax(const float* __restrict__ Q, const float* __restrict__ K,
                   const float* __restrict__ V, unsigned short* __restrict__ Ctx)
{
    __shared__ float kk[1024];
    __shared__ float vv[1024];

    const int    tid  = threadIdx.x;
    const size_t base = (size_t)blockIdx.x * 1024;

    *reinterpret_cast<float4*>(&kk[tid * 4]) =
        *reinterpret_cast<const float4*>(K + base + tid * 4);
    *reinterpret_cast<float4*>(&vv[tid * 4]) =
        *reinterpret_cast<const float4*>(V + base + tid * 4);
    __syncthreads();

    const float4 qv = *reinterpret_cast<const float4*>(Q + base + tid * 4);
    const int    hb = (tid >> 4) << 6;          // head base (64 ch/head)
    const float  q4[4] = {qv.x, qv.y, qv.z, qv.w};
    float        out[4];

    #pragma unroll
    for (int c = 0; c < 4; ++c) {
        const float s = q4[c] * 0.125f;         // 1/sqrt(64)
        float mx = -__builtin_inff();
        #pragma unroll 8
        for (int j = 0; j < 64; ++j) mx = fmaxf(mx, s * kk[hb + j]);
        float se = 0.f, av = 0.f;
        #pragma unroll 8
        for (int j = 0; j < 64; ++j) {
            const float e = __expf(s * kk[hb + j] - mx);   // v_exp_f32
            se += e;
            av = fmaf(e, vv[hb + j], av);
        }
        out[c] = av / se;
    }
    ushort4 h;
    h.x = f2bf(out[0]); h.y = f2bf(out[1]); h.z = f2bf(out[2]); h.w = f2bf(out[3]);
    *reinterpret_cast<ushort4*>(Ctx + base + tid * 4) = h;
}

extern "C" void kernel_launch(void* const* d_in, const int* in_sizes, int n_in,
                              void* d_out, int out_size, void* d_ws, size_t ws_size,
                              hipStream_t stream) {
    (void)in_sizes; (void)n_in; (void)out_size; (void)ws_size;
    const float* x_q = (const float*)d_in[0];
    const float* x_k = (const float*)d_in[1];
    const float* x_v = (const float*)d_in[2];
    const float* Wq  = (const float*)d_in[3];
    const float* bq  = (const float*)d_in[4];
    const float* Wk  = (const float*)d_in[5];
    const float* bk  = (const float*)d_in[6];
    const float* Wv  = (const float*)d_in[7];
    const float* bv  = (const float*)d_in[8];
    const float* Wo  = (const float*)d_in[9];
    const float* bo  = (const float*)d_in[10];
    float* out = (float*)d_out;

    const int Mrows = 4 * 512;                    // 2048 tokens
    const int D     = 1024;
    const size_t actElems = (size_t)Mrows * D;    // 2M
    const size_t wElems   = (size_t)D * D;        // 1M

    // workspace layout (48 MB total)
    unsigned short* bXq = (unsigned short*)d_ws;
    unsigned short* bXk = bXq + actElems;
    unsigned short* bXv = bXk + actElems;
    unsigned short* bWq = bXv + actElems;
    unsigned short* bWk = bWq + wElems;
    unsigned short* bWv = bWk + wElems;
    unsigned short* bWo = bWv + wElems;
    float*          Qf  = (float*)(bWo + wElems);
    float*          Kf  = Qf + actElems;
    float*          Vf  = Kf + actElems;
    unsigned short* Cx  = (unsigned short*)(Vf + actElems);

    dim3 block(256);
    const int actB = (int)(actElems / 4 / 256);   // 2048 blocks
    const int wB   = (int)(wElems   / 4 / 256);   // 1024 blocks

    cvt_f32_bf16<<<actB, block, 0, stream>>>(x_q, bXq, (int)(actElems / 4));
    cvt_f32_bf16<<<actB, block, 0, stream>>>(x_k, bXk, (int)(actElems / 4));
    cvt_f32_bf16<<<actB, block, 0, stream>>>(x_v, bXv, (int)(actElems / 4));
    cvt_f32_bf16<<<wB,   block, 0, stream>>>(Wq,  bWq, (int)(wElems / 4));
    cvt_f32_bf16<<<wB,   block, 0, stream>>>(Wk,  bWk, (int)(wElems / 4));
    cvt_f32_bf16<<<wB,   block, 0, stream>>>(Wv,  bWv, (int)(wElems / 4));
    cvt_f32_bf16<<<wB,   block, 0, stream>>>(Wo,  bWo, (int)(wElems / 4));

    dim3 gridG(D / BN, Mrows / BM);               // (8, 16)
    gemm_bias_bf16wmma<<<gridG, block, 0, stream>>>(bXq, bWq, bq, Qf, Mrows, D, D);
    gemm_bias_bf16wmma<<<gridG, block, 0, stream>>>(bXk, bWk, bk, Kf, Mrows, D, D);
    gemm_bias_bf16wmma<<<gridG, block, 0, stream>>>(bXv, bWv, bv, Vf, Mrows, D, D);

    attn_rank1_softmax<<<dim3(Mrows), block, 0, stream>>>(Qf, Kf, Vf, Cx);

    gemm_bias_bf16wmma<<<gridG, block, 0, stream>>>(Cx, bWo, bo, out, Mrows, D, D);
}
